// MoEBlock_5592047420171
// MI455X (gfx1250) — matrix operand order repaired and checked by reading, not compiled, not run on hardware
//
#include <hip/hip_runtime.h>

typedef _Float16 f16;
typedef __attribute__((ext_vector_type(16))) _Float16 v16h;
typedef __attribute__((ext_vector_type(8)))  _Float16 v8h;
typedef __attribute__((ext_vector_type(8)))  float    v8f;
typedef __attribute__((ext_vector_type(4)))  float    v4f;

#define T_TOK 8192   // B*S
#define DM    1024
#define DF    4096
#define NE    8
#define NR    16

#define WMMA_F16(a, b, c) \
  __builtin_amdgcn_wmma_f32_16x16x32_f16(false, (a), false, (b), (short)0, (c), false, false)

// ---- fragment loaders (CDNA5 WMMA 16x16x32 f16 layouts) --------------------
// A (16x32, M=row, K=col), row-major source with row stride `stride` halfs:
// lane: row = lane&15, kh = lane>>4; elems 0..7 -> K = 8*kh + j, 8..15 -> K = 16 + 8*kh + j
__device__ __forceinline__ v16h load_fragA_rm(const f16* __restrict__ base, int stride) {
  const int lane = threadIdx.x & 31;
  const int row  = lane & 15;
  const int kh   = lane >> 4;
  const f16* p = base + (size_t)row * stride + kh * 8;
  v8h lo = *(const v8h*)(p);
  v8h hi = *(const v8h*)(p + 16);
  v16h r;
#pragma unroll
  for (int i = 0; i < 8; ++i) { r[i] = lo[i]; r[i + 8] = hi[i]; }
  return r;
}

// B (32x16, K=row, N=col), source row-major with N as rows (wi/wo store [N][K]):
// lane: col = lane&15, kh = lane>>4; elems 0..7 -> K = 8*kh + j, 8..15 -> K = 16 + 8*kh + j
__device__ __forceinline__ v16h load_fragB_rm(const f16* __restrict__ base, int stride) {
  const int lane = threadIdx.x & 31;
  const int col  = lane & 15;
  const int kh   = lane >> 4;
  const f16* p = base + (size_t)col * stride + kh * 8;
  v8h lo = *(const v8h*)(p);
  v8h hi = *(const v8h*)(p + 16);
  v16h r;
#pragma unroll
  for (int i = 0; i < 8; ++i) { r[i] = lo[i]; r[i + 8] = hi[i]; }
  return r;
}

// ---- fp32 -> fp16 convert ---------------------------------------------------
__global__ void cvt_f32_f16(const float* __restrict__ in, f16* __restrict__ out, int n) {
  int i = blockIdx.x * blockDim.x + threadIdx.x;
  if (i < n) out[i] = (f16)in[i];
}

// ---- router: logits -> softmax -> top-2 -> per-expert gate coef -------------
__global__ void router_kernel(const float* __restrict__ X, const float* __restrict__ gw,
                              const float* __restrict__ gb, float* __restrict__ gates,
                              float* __restrict__ gsum) {
  const int wave = threadIdx.x >> 5;
  const int lane = threadIdx.x & 31;
  const int t = blockIdx.x * 8 + wave;
  float part[NE];
#pragma unroll
  for (int e = 0; e < NE; ++e) part[e] = 0.f;
  const float* xr = X + (size_t)t * DM;
  for (int i = lane; i < DM; i += 32) {
    float x = xr[i];
#pragma unroll
    for (int e = 0; e < NE; ++e) part[e] += x * gw[e * DM + i];
  }
#pragma unroll
  for (int e = 0; e < NE; ++e) {
#pragma unroll
    for (int off = 16; off > 0; off >>= 1)
      part[e] += __shfl_xor(part[e], off, 32);
  }
  if (lane == 0) {
    float lg[NE], mx = -1e30f;
    for (int e = 0; e < NE; ++e) { lg[e] = part[e] + gb[e]; mx = fmaxf(mx, lg[e]); }
    float s = 0.f;
    for (int e = 0; e < NE; ++e) { lg[e] = __expf(lg[e] - mx); s += lg[e]; }
    float inv = 1.f / s;
    for (int e = 0; e < NE; ++e) lg[e] *= inv;
    int i1 = 0;
    for (int e = 1; e < NE; ++e) if (lg[e] > lg[i1]) i1 = e;
    int i2 = (i1 == 0) ? 1 : 0;
    for (int e = 0; e < NE; ++e) if (e != i1 && lg[e] > lg[i2]) i2 = e;
    float vs = lg[i1] + lg[i2];
    for (int e = 0; e < NE; ++e) gates[(size_t)t * NE + e] = (e == i1 || e == i2) ? vs : 0.f;
    gsum[t] = 2.f * vs;  // sum over experts of count_e * val_sum
  }
}

// ---- P[t][e][r] = X[t,:] . lora_A[e][r,:]  (stored f16) ---------------------
// Stages the token row into LDS with CDNA5 async copies (ASYNCcnt path).
__global__ void lora_proj_kernel(const float* __restrict__ X, const float* __restrict__ A,
                                 f16* __restrict__ P) {
  __shared__ float xrow[DM];
  const int t = blockIdx.x;
#pragma unroll
  for (int it = 0; it < 2; ++it) {
    int idx = threadIdx.x + it * 128;  // 256 chunks of 16B = 4KB row
    unsigned lds = (unsigned)(size_t)(&xrow[idx * 4]);
    unsigned long long ga = (unsigned long long)(X + (size_t)t * DM + idx * 4);
    asm volatile("global_load_async_to_lds_b128 %0, %1, off"
                 :: "v"(lds), "v"(ga) : "memory");
  }
  asm volatile("s_wait_asynccnt 0" ::: "memory");
  __syncthreads();
  const int e = threadIdx.x >> 4;
  const int r = threadIdx.x & 15;
  const float* ar = A + ((size_t)e * NR + r) * DM;
  float s = 0.f;
#pragma unroll 8
  for (int d = 0; d < DM; ++d) s += xrow[d] * ar[d];
  P[(size_t)t * (NE * NR) + e * NR + r] = (f16)s;
}

// ---- fused up-proj + LoRA(all 8 experts) + ReLU + gated sum -> Hh -----------
// H[t,f] = sum_e gate[t,e] * relu( (X wi^T)[t,f] + bi[f] + (P_e B_e^T)[t,f] )
// Wave computes a 64x32 register tile (4x2 WMMA tiles); block = 128 tok x 128 f.
__global__ void __launch_bounds__(256)
moe_up_kernel(const f16* __restrict__ Xh, const f16* __restrict__ wih,
              const float* __restrict__ bi, const f16* __restrict__ Ph,
              const f16* __restrict__ Bh, const float* __restrict__ gates,
              f16* __restrict__ Hh) {
  __shared__ float gl[128 * NE];  // gates for the 128-token block tile (4KB)
  __shared__ f16   tb[8 * 256];   // per-wave 16x16 transpose buffer
  const int wave = threadIdx.x >> 5;
  const int lane = threadIdx.x & 31;
  const int n    = lane & 15;
  const int kh   = lane >> 4;
  const int mw   = wave >> 2;      // 2 M-groups of waves (64 tokens each)
  const int nw   = wave & 3;       // 4 N-groups of waves (32 cols each)
  const int tt = blockIdx.x >> 5;  // 64 token blocks (128 tokens each)
  const int fb = blockIdx.x & 31;  // 32 f blocks (128 cols each)
  const int t0b = tt * 128;
  const int t0  = t0b + mw * 64;
  const int f0  = fb * 128 + nw * 32;

#pragma unroll
  for (int i = 0; i < 4; ++i)
    gl[threadIdx.x + i * 256] = gates[(size_t)t0b * NE + threadIdx.x + i * 256];
  __syncthreads();

  // base = X @ wi^T over K = 1024, 4x2 register blocking
  v8f acc[4][2] = {};
  const f16* ab[4];
#pragma unroll
  for (int mi = 0; mi < 4; ++mi) ab[mi] = Xh + (size_t)(t0 + mi * 16) * DM;
  const f16* b0b = wih + (size_t)f0 * DM;
  const f16* b1b = wih + (size_t)(f0 + 16) * DM;
  for (int kt = 0; kt < DM / 32; ++kt) {
    v16h a[4];
#pragma unroll
    for (int mi = 0; mi < 4; ++mi) a[mi] = load_fragA_rm(ab[mi] + kt * 32, DM);
    v16h b0 = load_fragB_rm(b0b + kt * 32, DM);
    v16h b1 = load_fragB_rm(b1b + kt * 32, DM);
#pragma unroll
    for (int mi = 0; mi < 4; ++mi) {
      acc[mi][0] = WMMA_F16(a[mi], b0, acc[mi][0]);
      acc[mi][1] = WMMA_F16(a[mi], b1, acc[mi][1]);
    }
  }

  f16* tbw = tb + wave * 256;
#pragma unroll
  for (int mi = 0; mi < 4; ++mi) {
#pragma unroll
    for (int ni = 0; ni < 2; ++ni) {
      const int f    = f0 + ni * 16 + n;           // this lane's column
      const int trow = mw * 64 + mi * 16;          // tile row within block
      const float bif = bi[f];
      v8f basec;
#pragma unroll
      for (int v = 0; v < 8; ++v) basec[v] = acc[mi][ni][v] + bif;

      v8f hacc = {};
      for (int e = 0; e < NE; ++e) {
        // A = P_e tile (16 tokens x R=16, K padded to 32 with zeros)
        const v8h pv = *(const v8h*)(Ph + (size_t)(t0 + mi * 16 + n) * (NE * NR)
                                        + e * NR + kh * 8);
        // B = B_e tile (K=r rows x 16 f cols); elems 8..15 (K=16..31) -> zero
        const f16* bp = Bh + ((size_t)e * DF + f) * NR;
        const v8h bv0 = *(const v8h*)(bp);
        const v8h bv1 = *(const v8h*)(bp + 8);
        v16h pA, bB;
#pragma unroll
        for (int i = 0; i < 8; ++i) {
          pA[i]     = pv[i];
          pA[i + 8] = (f16)0.f;                    // K >= 16 of A
          bB[i]     = kh ? bv1[i] : bv0[i];        // K = 8*kh + i
          bB[i + 8] = (f16)0.f;
        }
        v8f inter = WMMA_F16(pA, bB, basec);
#pragma unroll
        for (int v = 0; v < 8; ++v) {
          float g = gl[(trow + kh * 8 + v) * NE + e];
          hacc[v] += g * fmaxf(inter[v], 0.f);
        }
      }

      // transpose D fragment (column-per-lane) through per-wave LDS buffer;
      // LDS ops are in-order within a wave, no block barrier needed.
#pragma unroll
      for (int v = 0; v < 8; ++v) tbw[(kh * 8 + v) * 16 + n] = (f16)hacc[v];
      v8h hv = *(const v8h*)(tbw + n * 16 + kh * 8);
      *(v8h*)(Hh + (size_t)(t0 + mi * 16 + n) * DF + f0 + ni * 16 + kh * 8) = hv;
    }
  }
}

// ---- down-proj: out = H @ wo^T + gatesum * bo -------------------------------
// Wave computes a 64x32 register tile (4x2 WMMA tiles); block = 128 tok x 128 d.
__global__ void __launch_bounds__(256)
moe_down_kernel(const f16* __restrict__ Hh, const f16* __restrict__ woh,
                const float* __restrict__ bo, const float* __restrict__ gsum,
                float* __restrict__ out) {
  __shared__ float tb[8 * 256];    // per-wave 16x16 f32 transpose buffer
  const int wave = threadIdx.x >> 5;
  const int lane = threadIdx.x & 31;
  const int n  = lane & 15;
  const int kh = lane >> 4;
  const int mw = wave >> 2;
  const int nw = wave & 3;
  const int tt = blockIdx.x >> 3;  // 64 token blocks (128 tokens)
  const int db = blockIdx.x & 7;   // 8 d blocks (128 cols)
  const int t0 = tt * 128 + mw * 64;
  const int d0 = db * 128 + nw * 32;

  v8f acc[4][2] = {};
  const f16* ab[4];
#pragma unroll
  for (int mi = 0; mi < 4; ++mi) ab[mi] = Hh + (size_t)(t0 + mi * 16) * DF;
  const f16* b0b = woh + (size_t)d0 * DF;
  const f16* b1b = woh + (size_t)(d0 + 16) * DF;
  for (int kt = 0; kt < DF / 32; ++kt) {
    v16h a[4];
#pragma unroll
    for (int mi = 0; mi < 4; ++mi) a[mi] = load_fragA_rm(ab[mi] + kt * 32, DF);
    v16h b0 = load_fragB_rm(b0b + kt * 32, DF);
    v16h b1 = load_fragB_rm(b1b + kt * 32, DF);
#pragma unroll
    for (int mi = 0; mi < 4; ++mi) {
      acc[mi][0] = WMMA_F16(a[mi], b0, acc[mi][0]);
      acc[mi][1] = WMMA_F16(a[mi], b1, acc[mi][1]);
    }
  }

  float* tbw = tb + wave * 256;
#pragma unroll
  for (int mi = 0; mi < 4; ++mi) {
#pragma unroll
    for (int ni = 0; ni < 2; ++ni) {
      const int d = d0 + ni * 16 + n;
      const float bod = bo[d];
#pragma unroll
      for (int v = 0; v < 8; ++v) {
        int m = kh * 8 + v;
        tbw[m * 16 + n] = acc[mi][ni][v] + gsum[t0 + mi * 16 + m] * bod;
      }
      // per-wave buffer: wave-internal LDS ordering suffices
#pragma unroll
      for (int it = 0; it < 2; ++it) {
        int idx = lane + it * 32;    // 64 chunks of 4 floats = 16x16 tile
        int row = idx >> 2;
        int c4  = (idx & 3) * 4;
        v4f val = *(const v4f*)(tbw + row * 16 + c4);
        *(v4f*)(out + (size_t)(t0 + mi * 16 + row) * DM + d0 + ni * 16 + c4) = val;
      }
    }
  }
}

// ---- host entry -------------------------------------------------------------
extern "C" void kernel_launch(void* const* d_in, const int* in_sizes, int n_in,
                              void* d_out, int out_size, void* d_ws, size_t ws_size,
                              hipStream_t stream) {
  (void)in_sizes; (void)n_in; (void)out_size; (void)ws_size;
  const float* X  = (const float*)d_in[0];
  const float* wi = (const float*)d_in[1];
  const float* bi = (const float*)d_in[2];
  const float* wo = (const float*)d_in[3];
  const float* bo = (const float*)d_in[4];
  const float* lA = (const float*)d_in[5];
  const float* lB = (const float*)d_in[6];
  const float* gw = (const float*)d_in[7];
  const float* gb = (const float*)d_in[8];
  float* out = (float*)d_out;

  char* ws = (char*)d_ws;
  size_t o = 0;
  f16* Xh  = (f16*)(ws + o); o += (size_t)T_TOK * DM * 2;
  f16* wih = (f16*)(ws + o); o += (size_t)DF * DM * 2;
  f16* woh = (f16*)(ws + o); o += (size_t)DM * DF * 2;
  f16* Bh  = (f16*)(ws + o); o += (size_t)NE * DF * NR * 2;
  f16* Ph  = (f16*)(ws + o); o += (size_t)T_TOK * NE * NR * 2;
  float* gates = (float*)(ws + o); o += (size_t)T_TOK * NE * 4;
  float* gsum  = (float*)(ws + o); o += (size_t)T_TOK * 4;
  f16* Hh  = (f16*)(ws + o); // T_TOK * DF * 2 bytes

  cvt_f32_f16<<<(T_TOK * DM + 255) / 256, 256, 0, stream>>>(X, Xh, T_TOK * DM);
  cvt_f32_f16<<<(DF * DM + 255) / 256, 256, 0, stream>>>(wi, wih, DF * DM);
  cvt_f32_f16<<<(DM * DF + 255) / 256, 256, 0, stream>>>(wo, woh, DM * DF);
  cvt_f32_f16<<<(NE * DF * NR + 255) / 256, 256, 0, stream>>>(lB, Bh, NE * DF * NR);

  router_kernel<<<T_TOK / 8, 256, 0, stream>>>(X, gw, gb, gates, gsum);
  lora_proj_kernel<<<T_TOK, 128, 0, stream>>>(X, lA, Ph);

  moe_up_kernel<<<(T_TOK / 128) * (DF / 128), 256, 0, stream>>>(
      Xh, wih, bi, Ph, Bh, gates, Hh);
  moe_down_kernel<<<(T_TOK / 128) * (DM / 128), 256, 0, stream>>>(
      Hh, woh, bo, gsum, out);
}
